// SelfAttention_25314537242853
// MI455X (gfx1250) — matrix operand order repaired and checked by reading, not compiled, not run on hardware
//
#include <hip/hip_runtime.h>

// Self-attention forward, flash-attention style, f16 WMMA / f32 accumulate.
// B=2, T=2048, C=1024, H=16, D=64.

#define BATCH 2
#define SEQ   2048
#define EMBD  1024
#define HEADS 16
#define HDIM  64
#define BT    (BATCH * SEQ)     // 4096
#define C3    (3 * EMBD)        // 3072

typedef _Float16 f16;
typedef __attribute__((ext_vector_type(16))) _Float16 v16h;
typedef __attribute__((ext_vector_type(8)))  _Float16 v8h;
typedef __attribute__((ext_vector_type(4)))  _Float16 v4h;
typedef __attribute__((ext_vector_type(8)))  float    v8f;
typedef __attribute__((ext_vector_type(4)))  int      v4i;

// gfx1250 async load-to-LDS path (guarded; falls back to plain copies).
#if defined(__has_builtin)
#  if __has_builtin(__builtin_amdgcn_global_load_async_to_lds_b128) && \
      __has_builtin(__builtin_amdgcn_s_wait_asynccnt)
#    define HAVE_ASYNC_LDS 1
#  endif
#endif
#ifndef HAVE_ASYNC_LDS
#  define HAVE_ASYNC_LDS 0
#endif

#if HAVE_ASYNC_LDS
typedef __attribute__((address_space(1))) v4i gv4i;
typedef __attribute__((address_space(3))) v4i lv4i;
static __device__ __forceinline__ void async_copy16(const f16* g, f16* l) {
  __builtin_amdgcn_global_load_async_to_lds_b128((gv4i*)g, (lv4i*)l, 0, 0);
}
#endif

// ---------------------------------------------------------------------------
// WMMA fragment helpers (wave32; contiguous k = 16*(lane/16) + e, same for A and B)
// ---------------------------------------------------------------------------

// A-matrix 16x32 f16: lane L<16 -> row M=L, K = 0..15;
//                     lane L>=16 -> row M=L-16, K = 16..31
static __device__ __forceinline__ v16h wmma_load_a(const f16* tile, int ld) {
  const int lane = threadIdx.x & 31;
  const int row  = lane & 15;
  const int kb   = (lane >> 4) << 4;   // 0 or 16
  const f16* p = tile + row * ld;
  v16h a;
#pragma unroll
  for (int e = 0; e < 16; ++e) {
    a[e] = p[kb + e];
  }
  return a;
}

// B-matrix 32x16 f16 where B(kk,n) = src[n*ld + kk]  (src holds B^T row-major
// -> each lane reads 16 contiguous halves).
// lane L<16 -> col N=L, K=0..15 ; lane L>=16 -> col N=L-16, K=16..31
static __device__ __forceinline__ v16h wmma_load_bT(const f16* src, int ld) {
  const int lane = threadIdx.x & 31;
  const int n    = lane & 15;
  const int kk0  = (lane >> 4) << 4;   // 0 or 16
  const f16* p = src + (size_t)n * ld + kk0;
  v16h b;
#pragma unroll
  for (int e = 0; e < 16; ++e) b[e] = p[e];
  return b;
}

static __device__ __forceinline__ v8f wmma_f16(v16h a, v16h b, v8f c) {
  return __builtin_amdgcn_wmma_f32_16x16x32_f16(
      /*neg_a=*/false, a, /*neg_b=*/false, b,
      /*c_mod=*/(short)0, c, /*reuse_a=*/false, /*reuse_b=*/false);
}

// Stage a 128x32 f16 tile (rows rBlk..rBlk+127, cols k0..k0+31 of a row-major
// [*, ldx] matrix) into LDS. 256 threads, 2 x 16B per thread.
// Async (TDM-style, ASYNCcnt-tracked) on gfx1250 when available.
static __device__ __forceinline__ void stage_tile(const f16* __restrict__ M,
                                                  f16* lds, int rBlk,
                                                  int k0, int ldx) {
#pragma unroll
  for (int t = 0; t < 2; ++t) {
    const int id  = threadIdx.x + t * 256;   // 0..511
    const int row = id >> 2;                 // 0..127
    const int kc  = (id & 3) * 8;            // 0,8,16,24
    const f16* src = M + (size_t)(rBlk + row) * ldx + k0 + kc;
    f16* dst = lds + row * 32 + kc;
#if HAVE_ASYNC_LDS
    async_copy16(src, dst);
#else
    *(v8h*)dst = *(const v8h*)src;
#endif
  }
}

static __device__ __forceinline__ void stage_wait() {
#if HAVE_ASYNC_LDS
  __builtin_amdgcn_s_wait_asynccnt(0);
#endif
  __syncthreads();
}

// ---------------------------------------------------------------------------
// fp32 -> f16 conversion (vectorized, sizes are multiples of 4)
// ---------------------------------------------------------------------------
__global__ void cast_f32_to_f16(const float4* __restrict__ in,
                                v4h* __restrict__ out, int n4) {
  int i = blockIdx.x * blockDim.x + threadIdx.x;
  const int stride = gridDim.x * blockDim.x;
  for (; i < n4; i += stride) {
    const float4 v = in[i];
    v4h o;
    o[0] = (f16)v.x; o[1] = (f16)v.y; o[2] = (f16)v.z; o[3] = (f16)v.w;
    out[i] = o;
  }
}

// Fused cast + transpose: in f32 [rows][cols] -> out f16 [cols][rows].
// 32x32 tiles via LDS, block (32,8).
__global__ __launch_bounds__(256) void cast_transpose_f16(
    const float* __restrict__ in, f16* __restrict__ out, int rows, int cols) {
  __shared__ f16 tile[32][33];
  const int c0 = blockIdx.x * 32;
  const int r0 = blockIdx.y * 32;
  const int tx = threadIdx.x;
  const int ty = threadIdx.y;
#pragma unroll
  for (int i = 0; i < 32; i += 8)
    tile[ty + i][tx] = (f16)in[(size_t)(r0 + ty + i) * cols + c0 + tx];
  __syncthreads();
#pragma unroll
  for (int i = 0; i < 32; i += 8)
    out[(size_t)(c0 + ty + i) * rows + r0 + tx] = tile[tx][ty + i];
}

// ---------------------------------------------------------------------------
// GEMM core: C[128x128 block] = A[BTxK] @ Wt^T + bias, K = EMBD.
// A row-major [m][k]; Wt is the TRANSPOSED weight, row-major [n][k] -> both
// operands stage as contiguous async b128 copies into double-buffered LDS.
// 256 threads / 8 waves; wave tile 32x64. Epilogue differs per kernel.
// ---------------------------------------------------------------------------
#define GEMM_PROLOGUE()                                                     \
  __shared__ __align__(16) f16 ldsA[2][128 * 32];                           \
  __shared__ __align__(16) f16 ldsB[2][128 * 32];                           \
  const int lane = threadIdx.x & 31;                                        \
  const int wave = threadIdx.x >> 5;                                        \
  const int wm = wave & 3;                                                  \
  const int wn = wave >> 2;                                                 \
  const int mBlk = blockIdx.x * 128;                                        \
  const int mOff = mBlk + wm * 32;                                          \
  const int nBlk = blockIdx.y * 128;                                        \
  const int nOff = nBlk + wn * 64;                                          \
  v8f acc[2][4];                                                            \
  _Pragma("unroll") for (int i = 0; i < 2; ++i)                             \
      _Pragma("unroll") for (int j = 0; j < 4; ++j)                         \
      _Pragma("unroll") for (int r = 0; r < 8; ++r) acc[i][j][r] = 0.0f;    \
  int buf = 0;                                                              \
  stage_tile(X,  ldsA[0], mBlk, 0, EMBD);                                   \
  stage_tile(Wt, ldsB[0], nBlk, 0, EMBD);                                   \
  stage_wait();                                                             \
  for (int k0 = 0; k0 < EMBD; k0 += 32) {                                   \
    const int nxt = buf ^ 1;                                                \
    if (k0 + 32 < EMBD) {                                                   \
      stage_tile(X,  ldsA[nxt], mBlk, k0 + 32, EMBD);                       \
      stage_tile(Wt, ldsB[nxt], nBlk, k0 + 32, EMBD);                       \
    }                                                                       \
    v16h a[2];                                                              \
    _Pragma("unroll") for (int i = 0; i < 2; ++i)                           \
        a[i] = wmma_load_a(ldsA[buf] + (wm * 32 + 16 * i) * 32, 32);        \
    _Pragma("unroll") for (int j = 0; j < 4; ++j) {                         \
      v16h b = wmma_load_bT(ldsB[buf] + (wn * 64 + j * 16) * 32, 32);       \
      _Pragma("unroll") for (int i = 0; i < 2; ++i)                         \
          acc[i][j] = wmma_f16(a[i], b, acc[i][j]);                         \
    }                                                                       \
    stage_wait();                                                           \
    buf = nxt;                                                              \
  }

// QKV GEMM: Q,K written [B,H,T,D]; V written TRANSPOSED [B,H,D,T].
__global__ __launch_bounds__(256) void gemm_qkv(
    const f16* __restrict__ X, const f16* __restrict__ Wt,
    const float* __restrict__ bias,
    f16* __restrict__ Qm, f16* __restrict__ Km, f16* __restrict__ Vt) {
  GEMM_PROLOGUE()

  const int nl = lane & 15, hi = lane >> 4;
#pragma unroll
  for (int i = 0; i < 2; ++i)
#pragma unroll
    for (int j = 0; j < 4; ++j) {
      const int n = nOff + j * 16 + nl;
      const int which = n >> 10;              // 0=Q 1=K 2=V
      const int c = n & (EMBD - 1);
      const int h = c >> 6;
      const int d = c & 63;
      const float bv = bias[n];
#pragma unroll
      for (int r = 0; r < 8; ++r) {
        const int m  = mOff + i * 16 + r + 8 * hi;
        const int bb = m >> 11;               // / SEQ
        const int t  = m & (SEQ - 1);
        const f16 val = (f16)(acc[i][j][r] + bv);
        if (which == 2) {
          Vt[((size_t)(bb * HEADS + h) * HDIM + d) * SEQ + t] = val;
        } else {
          f16* dst = (which == 0) ? Qm : Km;
          dst[((size_t)(bb * HEADS + h) * SEQ + t) * HDIM + d] = val;
        }
      }
    }
}

// Output projection: fp32 out + bias.
__global__ __launch_bounds__(256) void gemm_proj(
    const f16* __restrict__ X, const f16* __restrict__ Wt,
    const float* __restrict__ bias, float* __restrict__ out) {
  GEMM_PROLOGUE()

  const int nl = lane & 15, hi = lane >> 4;
#pragma unroll
  for (int i = 0; i < 2; ++i)
#pragma unroll
    for (int j = 0; j < 4; ++j) {
      const int n = nOff + j * 16 + nl;
      const float bv = bias[n];
#pragma unroll
      for (int r = 0; r < 8; ++r) {
        const int m = mOff + i * 16 + r + 8 * hi;
        out[(size_t)m * EMBD + n] = acc[i][j][r] + bv;
      }
    }
}

// ---------------------------------------------------------------------------
// Flash attention: one wave per 32-row query tile per (b,h).
// S = Q K^T (scaled, causal), online softmax, O += P V, O /= l.
// K is [B,H,T,D] (KT fragments contiguous); V is [B,H,D,T] (PV fragments
// contiguous). Writes Y in [B, T, C] f16 for the proj GEMM.
// ---------------------------------------------------------------------------
__global__ __launch_bounds__(32) void attn(
    const f16* __restrict__ Q, const f16* __restrict__ K,
    const f16* __restrict__ Vt, f16* __restrict__ Y) {
  __shared__ __align__(16) f16 P[32 * 32];

  const int bh = blockIdx.x >> 6;          // / (SEQ/32)
  const int qt = blockIdx.x & 63;
  const int q0 = qt * 32;
  const f16* Qb = Q  + (size_t)bh * SEQ * HDIM;
  const f16* Kb = K  + (size_t)bh * SEQ * HDIM;
  const f16* Vb = Vt + (size_t)bh * HDIM * SEQ;

  const int lane = threadIdx.x & 31;
  const int nl = lane & 15, hi = lane >> 4;
  const float NEG_INF = -__builtin_inff();
  const float att_scale = 0.125f;          // 1/sqrt(64)

  // Q tile A-fragments: 2 row-tiles x 2 K-chunks, resident in VGPRs.
  v16h aq[2][2];
#pragma unroll
  for (int i = 0; i < 2; ++i)
#pragma unroll
    for (int dc = 0; dc < 2; ++dc)
      aq[i][dc] = wmma_load_a(Qb + (size_t)(q0 + 16 * i) * HDIM + dc * 32, HDIM);

  v8f accO[2][4];
#pragma unroll
  for (int i = 0; i < 2; ++i)
#pragma unroll
    for (int dt = 0; dt < 4; ++dt)
#pragma unroll
      for (int r = 0; r < 8; ++r) accO[i][dt][r] = 0.0f;

  float mrow[2][8], lrow[2][8];
#pragma unroll
  for (int i = 0; i < 2; ++i)
#pragma unroll
    for (int r = 0; r < 8; ++r) { mrow[i][r] = NEG_INF; lrow[i][r] = 0.0f; }

  for (int kb = 0; kb < q0 + 32; kb += 32) {   // 32-key chunks
    // ---- S = Q K^T for both row-tiles ----
    v8f s[2][2];
#pragma unroll
    for (int i = 0; i < 2; ++i)
#pragma unroll
      for (int j = 0; j < 2; ++j)
#pragma unroll
        for (int r = 0; r < 8; ++r) s[i][j][r] = 0.0f;
#pragma unroll
    for (int j = 0; j < 2; ++j)
#pragma unroll
      for (int dc = 0; dc < 2; ++dc) {
        v16h bk = wmma_load_bT(Kb + (size_t)(kb + 16 * j) * HDIM + dc * 32,
                               HDIM);
#pragma unroll
        for (int i = 0; i < 2; ++i) s[i][j] = wmma_f16(aq[i][dc], bk, s[i][j]);
      }

    // ---- causal mask + online softmax per row ----
    float pscale[2][8];
#pragma unroll
    for (int i = 0; i < 2; ++i)
#pragma unroll
      for (int r = 0; r < 8; ++r) {
        const int qrow = q0 + 16 * i + r + 8 * hi;
        const int key0 = kb + nl;
        float s0 = (key0 > qrow)      ? NEG_INF : s[i][0][r] * att_scale;
        float s1 = (key0 + 16 > qrow) ? NEG_INF : s[i][1][r] * att_scale;
        float mx = fmaxf(s0, s1);
#pragma unroll
        for (int xm = 1; xm < 16; xm <<= 1)
          mx = fmaxf(mx, __shfl_xor(mx, xm, 32));
        const float mnew = fmaxf(mrow[i][r], mx);
        const float corr = __expf(mrow[i][r] - mnew);
        const float p0 = __expf(s0 - mnew);
        const float p1 = __expf(s1 - mnew);
        float rs = p0 + p1;
#pragma unroll
        for (int xm = 1; xm < 16; xm <<= 1) rs += __shfl_xor(rs, xm, 32);
        lrow[i][r] = lrow[i][r] * corr + rs;
        mrow[i][r] = mnew;
        pscale[i][r] = corr;
        const int row = 16 * i + r + 8 * hi;
        P[row * 32 + nl]      = (f16)p0;     // D-layout -> LDS tile
        P[row * 32 + 16 + nl] = (f16)p1;
      }
#pragma unroll
    for (int i = 0; i < 2; ++i)
#pragma unroll
      for (int dt = 0; dt < 4; ++dt)
#pragma unroll
        for (int r = 0; r < 8; ++r) accO[i][dt][r] *= pscale[i][r];

    __syncthreads();                      // LDS visibility within wave
    v16h ap[2];
#pragma unroll
    for (int i = 0; i < 2; ++i) ap[i] = wmma_load_a(P + 16 * i * 32, 32);
#pragma unroll
    for (int dt = 0; dt < 4; ++dt) {
      v16h bv = wmma_load_bT(Vb + (size_t)(dt * 16) * SEQ + kb, SEQ);
#pragma unroll
      for (int i = 0; i < 2; ++i)
        accO[i][dt] = wmma_f16(ap[i], bv, accO[i][dt]);
    }
    __syncthreads();
  }

  // Epilogue: O /= l, write Y[b, t, h*64 + d] as f16.
  const int b = bh >> 4, h = bh & 15;
#pragma unroll
  for (int i = 0; i < 2; ++i)
#pragma unroll
    for (int dt = 0; dt < 4; ++dt)
#pragma unroll
      for (int r = 0; r < 8; ++r) {
        const int row = q0 + 16 * i + r + 8 * hi;
        const float ov = accO[i][dt][r] / lrow[i][r];
        Y[((size_t)(b * SEQ + row)) * EMBD + h * HDIM + dt * 16 + nl] = (f16)ov;
      }
}

// ---------------------------------------------------------------------------
// Host-side launch
// ---------------------------------------------------------------------------
extern "C" void kernel_launch(void* const* d_in, const int* in_sizes, int n_in,
                              void* d_out, int out_size, void* d_ws,
                              size_t ws_size, hipStream_t stream) {
  const float* x      = (const float*)d_in[0];  // [B,T,C]
  const float* W_qkv  = (const float*)d_in[1];  // [C,3C]
  const float* b_qkv  = (const float*)d_in[2];  // [3C]
  const float* W_proj = (const float*)d_in[3];  // [C,C]
  const float* b_proj = (const float*)d_in[4];  // [C]
  float* out = (float*)d_out;

  // Workspace carve-up (all f16):
  //   xh [BT,C]  wqkvT [3C,C]  wprojT [C,C]  Q/K [B,H,T,D]  Vt [B,H,D,T]
  //   Yh [BT,C]
  char* ws = (char*)d_ws;
  auto take = [&](size_t bytes) {
    char* p = ws;
    ws += (bytes + 255) & ~(size_t)255;
    return p;
  };
  f16* xh     = (f16*)take((size_t)BT * EMBD * sizeof(f16));
  f16* wqkvT  = (f16*)take((size_t)EMBD * C3 * sizeof(f16));
  f16* wprojT = (f16*)take((size_t)EMBD * EMBD * sizeof(f16));
  f16* Qm     = (f16*)take((size_t)BT * EMBD * sizeof(f16));
  f16* Km     = (f16*)take((size_t)BT * EMBD * sizeof(f16));
  f16* Vt     = (f16*)take((size_t)BT * EMBD * sizeof(f16));
  f16* Yh     = (f16*)take((size_t)BT * EMBD * sizeof(f16));

  {  // x: plain cast
    const int n4 = (BT * EMBD) / 4;
    int blocks = (n4 + 255) / 256;
    if (blocks > 2048) blocks = 2048;
    cast_f32_to_f16<<<blocks, 256, 0, stream>>>((const float4*)x, (v4h*)xh, n4);
  }
  // Weights: fused cast + transpose ([K][N] f32 -> [N][K] f16).
  cast_transpose_f16<<<dim3(C3 / 32, EMBD / 32), dim3(32, 8), 0, stream>>>(
      W_qkv, wqkvT, EMBD, C3);
  cast_transpose_f16<<<dim3(EMBD / 32, EMBD / 32), dim3(32, 8), 0, stream>>>(
      W_proj, wprojT, EMBD, EMBD);

  gemm_qkv<<<dim3(BT / 128, C3 / 128), 256, 0, stream>>>(
      xh, wqkvT, b_qkv, Qm, Km, Vt);

  attn<<<dim3(BATCH * HEADS * (SEQ / 32)), 32, 0, stream>>>(Qm, Km, Vt, Yh);

  gemm_proj<<<dim3(BT / 128, EMBD / 128), 256, 0, stream>>>(
      Yh, wprojT, b_proj, out);
}
